// RandomGRU_54322746359880
// MI455X (gfx1250) — matrix operand order repaired and checked by reading, not compiled, not run on hardware
//
#include <hip/hip_runtime.h>

typedef __attribute__((ext_vector_type(16))) _Float16 v16h;
typedef __attribute__((ext_vector_type(8)))  float    v8f;

#define TT 12
#define HH 64
#define MM 32
#define PPD 12
#define BN (64 * 2048)

// ---------------- workspace layout ----------------
// [0, 294912)          : WB fragments, f16, index = ((((t*3+g)*4+nb)*2+kb)*32 + lane)*16 + j
// [294912, 294912+36864): XW float4, index = (t*3+g)*64 + n : (w0, w1, bias, bch)
#define WS_WB_BYTES 294912

struct FragArgs { const float* Wl[3]; const float* bl[3]; };
struct XWArgs {
  const float *Wlx[3], *blx[3], *Wbx[3], *bbx[3], *Wbh[3], *bbh[3];
};

// Fast activations: v_exp_f32 + v_rcp_f32 (TRANS pipe) instead of IEEE div expansion.
__device__ __forceinline__ float fast_sigmoid(float x) {
  return __builtin_amdgcn_rcpf(1.0f + __expf(-x));
}
__device__ __forceinline__ float fast_tanh(float x) {
  return 2.0f * __builtin_amdgcn_rcpf(1.0f + __expf(-2.0f * x)) - 1.0f;
}

// Kernel A1: meta-generate hidden-gate weights directly in WMMA B-fragment layout.
// B (32x16, f16) k order used here, as in A: lanes 0-15 hold K=0..15 (half j -> K=j), lanes 16-31 hold
// K=16..31 (half j -> K=16+j); N = lane%16. B[k,n] = W[n,k].
__global__ __launch_bounds__(256) void meta_frags(const float* __restrict__ wz,
                                                  FragArgs a,
                                                  _Float16* __restrict__ wb) {
  int tid  = blockIdx.x * 256 + threadIdx.x;   // exactly 147456 threads
  int j    = tid & 15;
  int lane = (tid >> 4) & 31;
  int kb   = (tid >> 9) & 1;
  int nb   = (tid >> 10) & 3;
  int g    = (tid >> 12) % 3;
  int t    = tid / 12288;
  int lh   = lane & 15, hi = lane >> 4;
  int n    = nb * 16 + lh;
  int k    = kb * 32 + j + hi * 16;            // B-fragment K mapping
  int row  = n * HH + k;                       // Wl row = h*inn + i, inn = H
  const float* Wl  = a.Wl[g];
  const float* wzr = wz + t * MM;
  float acc = a.bl[g][row];
#pragma unroll
  for (int m = 0; m < MM; ++m) acc += wzr[m] * Wl[row * MM + m];
  wb[tid] = (_Float16)acc;
}

// Kernel A2: per-step x-side weights + combined biases.
__global__ __launch_bounds__(256) void meta_xw(const float* __restrict__ wz,
                                               const float* __restrict__ bz,
                                               XWArgs a,
                                               float4* __restrict__ XW) {
  int tid = blockIdx.x * 256 + threadIdx.x;
  if (tid >= TT * 3 * HH) return;
  int n = tid & 63;
  int g = (tid >> 6) % 3;
  int t = tid / 192;
  const float* wzr = wz + t * MM;
  const float* bzr = bz + t * MM;
  float w0 = a.blx[g][n * 2 + 0], w1 = a.blx[g][n * 2 + 1];
  float bx = a.bbx[g][n],          bh = a.bbh[g][n];
#pragma unroll
  for (int m = 0; m < MM; ++m) {
    w0 += wzr[m] * a.Wlx[g][(n * 2 + 0) * MM + m];
    w1 += wzr[m] * a.Wlx[g][(n * 2 + 1) * MM + m];
    bx += bzr[m] * a.Wbx[g][n * MM + m];
    bh += bzr[m] * a.Wbh[g][n * MM + m];
  }
  float4 o;
  o.x = w0; o.y = w1;
  if (g < 2) { o.z = bx + bh; o.w = 0.f; }   // r,z: fold both biases into C
  else       { o.z = bx;      o.w = bh;  }   // c: bcx outside, bch inside r*(...)
  XW[tid] = o;
}

// Kernel B: GRU over time with WMMA for the h @ W_h GEMMs.
// Per-wave-private LDS region: x tile (1536B) | h f16 [k][m] (2048B) | seq (768B).
__global__ __launch_bounds__(256) void gru_main(const float* __restrict__ x,
                                                const _Float16* __restrict__ WB,
                                                const float4* __restrict__ XW,
                                                const float* __restrict__ Wo,
                                                const float* __restrict__ bo,
                                                const float* __restrict__ Wp,
                                                const float* __restrict__ bp,
                                                float* __restrict__ out) {
  constexpr int PER_WAVE = 4352;
  __shared__ __align__(16) unsigned char lds[8 * PER_WAVE];

  const int tid  = threadIdx.x;
  const int wave = tid >> 5, lane = tid & 31;
  const int lh   = lane & 15, hi = lane >> 4;
  const int rgbase = blockIdx.x * 128 + wave * 16;

  float*     xl   = (float*)(lds + wave * PER_WAVE);
  _Float16*  hl   = (_Float16*)(lds + wave * PER_WAVE + 1536);
  float*     seql = (float*)(lds + wave * PER_WAVE + 1536 + 2048);

  // load x tile: 16 rows x 24 floats = 384 contiguous floats
  {
    const float4* xg = (const float4*)(x + (size_t)rgbase * (TT * 2));
    float4* xs = (float4*)xl;
    xs[lane]      = xg[lane];
    xs[lane + 32] = xg[lane + 32];
    xs[lane + 64] = xg[lane + 64];
  }
  // zero f16 h tile (2048B)
  {
    uint4 z = {0u, 0u, 0u, 0u};
    uint4* hz = (uint4*)hl;
    hz[lane] = z; hz[lane + 32] = z; hz[lane + 64] = z; hz[lane + 96] = z;
  }

  float wo[4];
#pragma unroll
  for (int nb = 0; nb < 4; ++nb) wo[nb] = Wo[nb * 16 + lh];
  const float bo0 = bo[0];

  v8f hf[4];
#pragma unroll
  for (int nb = 0; nb < 4; ++nb) { v8f zz = {}; hf[nb] = zz; }

  const v16h* WBv = (const v16h*)WB;

  for (int t = 0; t < TT; ++t) {
    // A fragments (16x32 f16, M = lane%16): lanes 0-15 K=0-15; lanes 16-31 K=16-31
    v16h a0, a1;
#pragma unroll
    for (int j = 0; j < 16; ++j) {
      int kl = j + (hi ? 16 : 0);
      a0[j] = hl[kl * 16 + lh];
      a1[j] = hl[(kl + 32) * 16 + lh];
    }

    v8f rf[4], zf[4];
#pragma unroll
    for (int g = 0; g < 2; ++g) {
#pragma unroll
      for (int nb = 0; nb < 4; ++nb) {
        float4 w = XW[(t * 3 + g) * 64 + nb * 16 + lh];
        v8f cfr;
#pragma unroll
        for (int i = 0; i < 8; ++i) {
          int m = i + hi * 8;
          cfr[i] = xl[m * 24 + t * 2] * w.x + xl[m * 24 + t * 2 + 1] * w.y + w.z;
        }
        int base = (((t * 3 + g) * 4 + nb) * 2) * 32;
        v16h b0 = WBv[base + lane];
        v16h b1 = WBv[base + 32 + lane];
        v8f d = __builtin_amdgcn_wmma_f32_16x16x32_f16(false, a0, false, b0,
                                                       (short)0, cfr, false, false);
        d = __builtin_amdgcn_wmma_f32_16x16x32_f16(false, a1, false, b1,
                                                   (short)0, d, false, false);
#pragma unroll
        for (int i = 0; i < 8; ++i) d[i] = fast_sigmoid(d[i]);
        if (g == 0) rf[nb] = d; else zf[nb] = d;
      }
    }

    v8f cf[4];
#pragma unroll
    for (int nb = 0; nb < 4; ++nb) {
      float4 w = XW[(t * 3 + 2) * 64 + nb * 16 + lh];
      v8f bch;
#pragma unroll
      for (int i = 0; i < 8; ++i) bch[i] = w.w;
      int base = (((t * 3 + 2) * 4 + nb) * 2) * 32;
      v16h b0 = WBv[base + lane];
      v16h b1 = WBv[base + 32 + lane];
      v8f hp = __builtin_amdgcn_wmma_f32_16x16x32_f16(false, a0, false, b0,
                                                      (short)0, bch, false, false);
      hp = __builtin_amdgcn_wmma_f32_16x16x32_f16(false, a1, false, b1,
                                                  (short)0, hp, false, false);
#pragma unroll
      for (int i = 0; i < 8; ++i) {
        int m = i + hi * 8;
        float xc = xl[m * 24 + t * 2] * w.x + xl[m * 24 + t * 2 + 1] * w.y + w.z;
        cf[nb][i] = fast_tanh(xc + rf[nb][i] * hp[i]);
      }
    }

    // hn = c + z*(h - c); write f16 copy to LDS [k][m]; output-head partials
    float p[8];
#pragma unroll
    for (int i = 0; i < 8; ++i) p[i] = 0.f;
#pragma unroll
    for (int nb = 0; nb < 4; ++nb) {
      union { _Float16 h[8]; uint4 u; } pk;
#pragma unroll
      for (int i = 0; i < 8; ++i) {
        float hn = cf[nb][i] + zf[nb][i] * (hf[nb][i] - cf[nb][i]);
        hf[nb][i] = hn;
        pk.h[i] = (_Float16)hn;
        p[i] += hn * wo[nb];
      }
      *(uint4*)(hl + (nb * 16 + lh) * 16 + hi * 8) = pk.u;  // 16B aligned
    }
    // reduce across the 16-lane half (bit 4 untouched)
#pragma unroll
    for (int mask = 1; mask < 16; mask <<= 1) {
#pragma unroll
      for (int i = 0; i < 8; ++i) p[i] += __shfl_xor(p[i], mask, 32);
    }
    if (lh == 0) {
#pragma unroll
      for (int i = 0; i < 8; ++i)
        seql[(i + hi * 8) * TT + t] = fmaxf(p[i] + bo0, 0.f);
    }
  }

  // predictor: seq [16 x 12] @ Wp^T [12 x 12] + bp -> write both output copies
  const size_t half = (size_t)BN * PPD;
#pragma unroll
  for (int q = 0; q < 6; ++q) {
    int o = lane + 32 * q;          // 0..191
    int m = o / PPD, pp = o % PPD;
    float acc = bp[pp];
#pragma unroll
    for (int t = 0; t < TT; ++t) acc += seql[m * TT + t] * Wp[pp * TT + t];
    size_t oi = (size_t)(rgbase + m) * PPD + pp;
    out[oi] = acc;
    out[half + oi] = acc;
  }
}

extern "C" void kernel_launch(void* const* d_in, const int* in_sizes, int n_in,
                              void* d_out, int out_size, void* d_ws, size_t ws_size,
                              hipStream_t stream) {
  const float* x  = (const float*)d_in[0];
  const float* wz = (const float*)d_in[2];
  const float* bz = (const float*)d_in[3];
  const float* Wo = (const float*)d_in[28];
  const float* bo = (const float*)d_in[29];
  const float* Wp = (const float*)d_in[30];
  const float* bp = (const float*)d_in[31];

  _Float16* wb = (_Float16*)d_ws;
  float4*   XW = (float4*)((char*)d_ws + WS_WB_BYTES);

  // hidden-gate meta sources: rh=8.., zh=16.., ch=24..
  FragArgs fa;
  fa.Wl[0] = (const float*)d_in[8];  fa.bl[0] = (const float*)d_in[9];
  fa.Wl[1] = (const float*)d_in[16]; fa.bl[1] = (const float*)d_in[17];
  fa.Wl[2] = (const float*)d_in[24]; fa.bl[2] = (const float*)d_in[25];

  XWArgs xa;
  const int xg[3] = {4, 12, 20};   // rx, zx, cx groups
  const int hg[3] = {8, 16, 24};   // rh, zh, ch groups
  for (int g = 0; g < 3; ++g) {
    xa.Wlx[g] = (const float*)d_in[xg[g] + 0];
    xa.blx[g] = (const float*)d_in[xg[g] + 1];
    xa.Wbx[g] = (const float*)d_in[xg[g] + 2];
    xa.bbx[g] = (const float*)d_in[xg[g] + 3];
    xa.Wbh[g] = (const float*)d_in[hg[g] + 2];
    xa.bbh[g] = (const float*)d_in[hg[g] + 3];
  }

  meta_frags<<<576, 256, 0, stream>>>(wz, fa, wb);        // 147456 threads
  meta_xw<<<9, 256, 0, stream>>>(wz, bz, xa, XW);         // 2304 threads
  gru_main<<<BN / 128, 256, 0, stream>>>(x, wb, XW, Wo, bo, Wp, bp, (float*)d_out);
}
